// BlockRecurrentAttention_16724602651135
// MI455X (gfx1250) — matrix operand order repaired and checked by reading, not compiled, not run on hardware
//
#include <hip/hip_runtime.h>
#include <cstdint>
#include <cstddef>

// ---------------------------------------------------------------------------
// Problem constants (from reference)
// ---------------------------------------------------------------------------
#define BB    4
#define NN    1024
#define SS    512
#define DD    1024
#define HH    8
#define DHD   64
#define INNER 512          // H * DH
#define QK_SCALE 0.125f    // DH^-0.5
#define RMS_EPS  1e-8f

typedef __bf16 bf16_t;
typedef __attribute__((ext_vector_type(2)))  __bf16 v2bf;
typedef __attribute__((ext_vector_type(16))) __bf16 v16bf;
typedef __attribute__((ext_vector_type(8)))  float  v8f;

// ---------------------------------------------------------------------------
// Helpers — use native f32->bf16 conversion (RNE) instead of bit tricks.
// ---------------------------------------------------------------------------
__device__ __forceinline__ bf16_t f2b(float f) {
    return (__bf16)f;
}
__device__ __forceinline__ uint32_t pack2(float a, float b) {
    v2bf p;
    p[0] = (__bf16)a;
    p[1] = (__bf16)b;
    return __builtin_bit_cast(uint32_t, p);
}

union FragU { v16bf v; uint32_t u[8]; };

// A-matrix fragment (16x32 bf16, this code's contiguous k order, k = 16h + s):
// lane L (L<16): row M=L, K = {0..15}; lane L+16: K = {16..31}.
// K-pairs are contiguous -> compiler emits 2x ds_load_b128 per fragment.
__device__ __forceinline__ v16bf load_afrag(const bf16_t* base, int stride,
                                            int row, int k0, int lane) {
    const int kh = lane >> 4;
    const bf16_t* rp = base + (size_t)row * stride;
    FragU r;
#pragma unroll
    for (int v = 0; v < 8; ++v) {
        int kb = k0 + 16 * kh + 2 * v;
        r.u[v] = *(const uint32_t*)(rp + kb);
    }
    return r.v;
}

// B-matrix fragment from an N-major LDS tile: element (K, col) lives at
// base[col*stride + K].  Lane (col = N index, kh = lane>>4) needs
// K = k0+16*kh .. k0+16*kh+15, a contiguous 32-byte run -> 2x ds_load_b128.
__device__ __forceinline__ v16bf load_bfrag_p(const bf16_t* base, int stride,
                                              int col, int k0, int lane) {
    const bf16_t* p = base + (size_t)col * stride + k0 + 16 * (lane >> 4);
    FragU r;
    uint4 a = *(const uint4*)(p);
    uint4 b = *(const uint4*)(p + 8);
    r.u[0] = a.x; r.u[1] = a.y; r.u[2] = a.z; r.u[3] = a.w;
    r.u[4] = b.x; r.u[5] = b.y; r.u[6] = b.z; r.u[7] = b.w;
    return r.v;
}

__device__ __forceinline__ v8f wmma_bf16(v16bf a, v16bf b, v8f c) {
    return __builtin_amdgcn_wmma_f32_16x16x32_bf16(
        /*neg_a=*/false, a, /*neg_b=*/false, b,
        /*c_mod=*/(short)0, c, /*reuse_a=*/false, /*reuse_b=*/false);
}

template <int ACT>
__device__ __forceinline__ float act_fn(float x) {
    if constexpr (ACT == 1) {            // exact GELU
        return 0.5f * x * (1.0f + erff(x * 0.70710678118654752f));
    } else if constexpr (ACT == 2) {     // tanh
        return tanhf(x);
    } else if constexpr (ACT == 3) {     // sigmoid(x - 1)
        return 1.0f / (1.0f + __expf(-(x - 1.0f)));
    } else if constexpr (ACT == 4) {     // sigmoid(x + 1)
        return 1.0f / (1.0f + __expf(-(x + 1.0f)));
    } else {
        return x;
    }
}

// ---------------------------------------------------------------------------
// GEMM: C[M,N] = act(A[M,K](f32) @ B(bf16, packed [K/32][N][32]) + bias) + add
// Workgroup tile 128x128x32, 8 waves, each wave 32x64 (2x4 WMMA tiles).
// Register double-buffering: tile kt+1 is fetched into registers (raw, no
// conversion) while tile kt is computed, hiding global-load latency.
// ---------------------------------------------------------------------------
template <int ACT, bool HAS_BIAS, bool HAS_ADD>
__global__ __launch_bounds__(256) void gemm_bf16_kernel(
    const float* __restrict__ A, const bf16_t* __restrict__ Bp,
    float* __restrict__ C, const float* __restrict__ bias,
    const float* __restrict__ add, int M, int N, int K) {
    __shared__ bf16_t As[128][40];   // row-major M x K, padded
    __shared__ bf16_t Bs[128][40];   // N-major:  N x K, padded

    const int tid  = threadIdx.x;
    const int lane = tid & 31;
    const int wave = tid >> 5;
    const int mb = (wave >> 1) * 32;   // wave M offset in tile
    const int nb = (wave & 1) * 64;    // wave N offset in tile
    const int bm = blockIdx.y * 128;
    const int bn = blockIdx.x * 128;

    const v8f vzero = {0.f, 0.f, 0.f, 0.f, 0.f, 0.f, 0.f, 0.f};
    v8f acc[2][4];
#pragma unroll
    for (int mt = 0; mt < 2; ++mt)
#pragma unroll
        for (int nt = 0; nt < 4; ++nt) acc[mt][nt] = vzero;

    const int arow = tid >> 1;
    const int acol = (tid & 1) * 16;
    const int kTiles = K >> 5;

    float4 areg[4];
    uint4  breg[2];

    auto global_fetch = [&](int kt) {
        const float* ag = A + (size_t)(bm + arow) * K + kt * 32 + acol;
#pragma unroll
        for (int i = 0; i < 4; ++i) areg[i] = *(const float4*)(ag + 4 * i);
        const bf16_t* bg = Bp + (size_t)kt * N * 32 + (size_t)bn * 32;
#pragma unroll
        for (int i = 0; i < 2; ++i) {
            int idx = tid + i * 256;
            breg[i] = *(const uint4*)(bg + (size_t)(idx >> 2) * 32 + (idx & 3) * 8);
        }
    };
    auto lds_store = [&]() {
        uint32_t pk[8];
#pragma unroll
        for (int i = 0; i < 4; ++i) {
            pk[2 * i + 0] = pack2(areg[i].x, areg[i].y);
            pk[2 * i + 1] = pack2(areg[i].z, areg[i].w);
        }
        uint4 w0 = {pk[0], pk[1], pk[2], pk[3]};
        uint4 w1 = {pk[4], pk[5], pk[6], pk[7]};
        *(uint4*)&As[arow][acol]     = w0;
        *(uint4*)&As[arow][acol + 8] = w1;
#pragma unroll
        for (int i = 0; i < 2; ++i) {
            int idx = tid + i * 256;
            *(uint4*)&Bs[idx >> 2][(idx & 3) * 8] = breg[i];
        }
    };

    global_fetch(0);
    lds_store();
    __syncthreads();

    for (int kt = 0; kt < kTiles; ++kt) {
        const bool has_next = (kt + 1) < kTiles;
        if (has_next) global_fetch(kt + 1);   // overlap with WMMA below

        v16bf af0 = load_afrag(&As[0][0], 40, mb + (lane & 15), 0, lane);
        v16bf af1 = load_afrag(&As[0][0], 40, mb + 16 + (lane & 15), 0, lane);
#pragma unroll
        for (int nt = 0; nt < 4; ++nt) {
            v16bf bf_ = load_bfrag_p(&Bs[0][0], 40, nb + nt * 16 + (lane & 15),
                                     0, lane);
            acc[0][nt] = wmma_bf16(af0, bf_, acc[0][nt]);
            acc[1][nt] = wmma_bf16(af1, bf_, acc[1][nt]);
        }

        if (has_next) {
            __syncthreads();   // all waves done reading LDS
            lds_store();
            __syncthreads();   // new tile visible
        }
    }

    const int kh = lane >> 4;
    const int cl = lane & 15;
#pragma unroll
    for (int mt = 0; mt < 2; ++mt)
#pragma unroll
        for (int nt = 0; nt < 4; ++nt) {
            int row0 = bm + mb + mt * 16 + 8 * kh;
            int col  = bn + nb + nt * 16 + cl;
            size_t idx = (size_t)row0 * N + col;
#pragma unroll
            for (int v = 0; v < 8; ++v) {
                float r = acc[mt][nt][v];
                if (HAS_BIAS) r += bias[col];
                r = act_fn<ACT>(r);
                if (HAS_ADD) r += add[idx];
                C[idx] = r;
                idx += (size_t)N;
            }
        }
}

// ---------------------------------------------------------------------------
// Flash attention: per (b,h), 64-query-row tile; QK^T and PV via WMMA bf16.
// Q/K/V layout: [B,H,T,64] f32.  Out: [B,T,INNER] f32.
// S=Q@K^T: B-operand K-dim is dh -> natural [key][dh] LDS layout is already
// fragment-contiguous.  P@V: K-dim is key -> V stored transposed [dh][key].
// Next K/V tile is prefetched (global_prefetch_b8) while computing.
// ---------------------------------------------------------------------------
template <bool CAUSAL>
__global__ __launch_bounds__(128) void flash_kernel(
    const float* __restrict__ Q, const float* __restrict__ Kp,
    const float* __restrict__ Vp, float* __restrict__ Out, int Tq, int Tk) {
    __shared__ bf16_t qs[64][72];   // [qrow][dh]
    __shared__ bf16_t ks[64][72];   // [key][dh]   (natural)
    __shared__ bf16_t vs[64][72];   // [dh][key]   (transposed)
    __shared__ bf16_t ps[64][72];   // [qrow][key]

    const int tid  = threadIdx.x;
    const int lane = tid & 31;
    const int wave = tid >> 5;      // 0..3
    const int mb   = wave * 16;
    const int kh   = lane >> 4;
    const int cl   = lane & 15;

    const int bh = blockIdx.y;      // b*H + h
    const int qb = blockIdx.x * 64;
    const float* Qb = Q  + (size_t)bh * Tq * 64;
    const float* Kb = Kp + (size_t)bh * Tk * 64;
    const float* Vb = Vp + (size_t)bh * Tk * 64;

    // stage Q (f32 -> bf16)
    {
        int r = tid >> 1, c0 = (tid & 1) * 32;
        const float* qg = Qb + (size_t)(qb + r) * 64 + c0;
#pragma unroll
        for (int i = 0; i < 8; ++i) {
            float4 f = *(const float4*)(qg + 4 * i);
            *(uint32_t*)&qs[r][c0 + 4 * i]     = pack2(f.x, f.y);
            *(uint32_t*)&qs[r][c0 + 4 * i + 2] = pack2(f.z, f.w);
        }
    }
    __syncthreads();

    v16bf aq0 = load_afrag(&qs[0][0], 72, mb + cl, 0, lane);
    v16bf aq1 = load_afrag(&qs[0][0], 72, mb + cl, 32, lane);

    const v8f vzero = {0.f, 0.f, 0.f, 0.f, 0.f, 0.f, 0.f, 0.f};
    v8f o[4];
    float m_[8], l_[8];
#pragma unroll
    for (int nt = 0; nt < 4; ++nt) o[nt] = vzero;
#pragma unroll
    for (int v = 0; v < 8; ++v) { m_[v] = -1e30f; l_[v] = 0.f; }

    int nTiles = Tk >> 6;
    if (CAUSAL) nTiles = blockIdx.x + 1;   // Tq == Tk for causal path

    const int skey = tid >> 1, sd0 = (tid & 1) * 32;

    for (int j = 0; j < nTiles; ++j) {
        // stage K (natural, vectorized) and V (transposed) tiles
        {
            const float* kg = Kb + (size_t)(j * 64 + skey) * 64 + sd0;
            const float* vg = Vb + (size_t)(j * 64 + skey) * 64 + sd0;
#pragma unroll
            for (int i = 0; i < 8; ++i) {
                float4 f = *(const float4*)(kg + 4 * i);
                *(uint32_t*)&ks[skey][sd0 + 4 * i]     = pack2(f.x, f.y);
                *(uint32_t*)&ks[skey][sd0 + 4 * i + 2] = pack2(f.z, f.w);
                float4 g = *(const float4*)(vg + 4 * i);
                vs[sd0 + 4 * i + 0][skey] = f2b(g.x);
                vs[sd0 + 4 * i + 1][skey] = f2b(g.y);
                vs[sd0 + 4 * i + 2][skey] = f2b(g.z);
                vs[sd0 + 4 * i + 3][skey] = f2b(g.w);
            }
        }
        // prefetch next tile into cache while we compute this one
        if (j + 1 < nTiles) {
            const float* kg2 = Kb + (size_t)((j + 1) * 64 + skey) * 64 + sd0;
            const float* vg2 = Vb + (size_t)((j + 1) * 64 + skey) * 64 + sd0;
            __builtin_prefetch(kg2, 0, 3);
            __builtin_prefetch(vg2, 0, 3);
        }
        __syncthreads();

        // S = Q @ K^T  (contraction over dh = 64)
        v8f s[4];
#pragma unroll
        for (int nt = 0; nt < 4; ++nt) {
            s[nt] = vzero;
            v16bf b0 = load_bfrag_p(&ks[0][0], 72, nt * 16 + cl, 0, lane);
            s[nt] = wmma_bf16(aq0, b0, s[nt]);
            v16bf b1 = load_bfrag_p(&ks[0][0], 72, nt * 16 + cl, 32, lane);
            s[nt] = wmma_bf16(aq1, b1, s[nt]);
        }

        if (CAUSAL && j == (int)blockIdx.x) {
#pragma unroll
            for (int nt = 0; nt < 4; ++nt)
#pragma unroll
                for (int v = 0; v < 8; ++v) {
                    int qrow = qb + mb + v + 8 * kh;
                    int kcol = j * 64 + nt * 16 + cl;
                    if (kcol > qrow) s[nt][v] = -1e30f;
                }
        }

        // online softmax (rows live across 16 lanes of each half-wave)
#pragma unroll
        for (int v = 0; v < 8; ++v) {
            float mx = s[0][v];
#pragma unroll
            for (int nt = 1; nt < 4; ++nt) mx = fmaxf(mx, s[nt][v]);
#pragma unroll
            for (int msk = 1; msk <= 8; msk <<= 1)
                mx = fmaxf(mx, __shfl_xor(mx, msk, 32));
            float mn   = fmaxf(m_[v], mx);
            float corr = __expf(m_[v] - mn);
            float rs = 0.f;
#pragma unroll
            for (int nt = 0; nt < 4; ++nt) {
                float p = __expf(s[nt][v] - mn);
                s[nt][v] = p;
                rs += p;
            }
#pragma unroll
            for (int msk = 1; msk <= 8; msk <<= 1)
                rs += __shfl_xor(rs, msk, 32);
            l_[v] = l_[v] * corr + rs;
            m_[v] = mn;
#pragma unroll
            for (int nt = 0; nt < 4; ++nt) o[nt][v] *= corr;
        }

        // P -> LDS (bf16)
#pragma unroll
        for (int nt = 0; nt < 4; ++nt)
#pragma unroll
            for (int v = 0; v < 8; ++v)
                ps[mb + v + 8 * kh][nt * 16 + cl] = f2b(s[nt][v]);
        __syncthreads();

        // O += P @ V  (contraction over key = 64; V is [dh][key])
        v16bf ap0 = load_afrag(&ps[0][0], 72, mb + cl, 0, lane);
        v16bf ap1 = load_afrag(&ps[0][0], 72, mb + cl, 32, lane);
#pragma unroll
        for (int nt = 0; nt < 4; ++nt) {
            v16bf b0 = load_bfrag_p(&vs[0][0], 72, nt * 16 + cl, 0, lane);
            o[nt] = wmma_bf16(ap0, b0, o[nt]);
            v16bf b1 = load_bfrag_p(&vs[0][0], 72, nt * 16 + cl, 32, lane);
            o[nt] = wmma_bf16(ap1, b1, o[nt]);
        }
        __syncthreads();
    }

    // epilogue: out[b, qrow, h*64 + dh] = o / l
    const int b = bh >> 3, h = bh & 7;
#pragma unroll
    for (int nt = 0; nt < 4; ++nt) {
        int row0 = qb + mb + 8 * kh;
        int col  = nt * 16 + cl;
        size_t idx = ((size_t)(b * Tq + row0)) * INNER + h * 64 + col;
#pragma unroll
        for (int v = 0; v < 8; ++v) {
            Out[idx] = o[nt][v] / l_[v];
            idx += (size_t)INNER;
        }
    }
}

// ---------------------------------------------------------------------------
// Elementwise kernels
// ---------------------------------------------------------------------------
__global__ __launch_bounds__(256) void rmsnorm_kernel(
    const float* __restrict__ x, const float* __restrict__ g,
    float* __restrict__ out) {
    __shared__ float red[256];
    const int row = blockIdx.x;
    const size_t base = (size_t)row * DD;
    const int tid = threadIdx.x;
    float ss = 0.f;
    for (int i = tid; i < DD; i += 256) { float v = x[base + i]; ss += v * v; }
    red[tid] = ss;
    __syncthreads();
    for (int s = 128; s > 0; s >>= 1) {
        if (tid < s) red[tid] += red[tid + s];
        __syncthreads();
    }
    float inv = 1.0f / fmaxf(sqrtf(red[0]) * 0.03125f, RMS_EPS); // *D^-0.5
    for (int i = tid; i < DD; i += 256)
        out[base + i] = x[base + i] * inv * g[i];
}

// Reshape [B*T, sstride] slice -> [B,H,T,64] with optional rope + scale.
__global__ __launch_bounds__(256) void prep_heads_kernel(
    const float* __restrict__ src, int sstride, float* __restrict__ dst,
    int T, float scale, int use_rope, int total) {
    int idx = blockIdx.x * 256 + threadIdx.x;
    if (idx >= total) return;
    int d   = idx & 63;
    int t64 = idx >> 6;
    int i   = t64 % T;
    int ht  = t64 / T;
    int h   = ht & 7;
    int b   = ht >> 3;
    size_t srow = (size_t)(b * T + i) * sstride + h * 64;
    float val;
    if (use_rope && d < 32) {
        int jj = d & 15;
        // inv_freq = 10000^{-jj/16};  ln(10000)/16 = 0.575646273248511
        float fr = (float)i * __expf(-0.575646273248511f * (float)jj);
        float cf = __cosf(fr), sf = __sinf(fr);
        float a  = src[srow + jj];
        float bq = src[srow + jj + 16];
        val = (d < 16) ? (a * cf - bq * sf) : (bq * cf + a * sf);
    } else {
        val = src[srow + d];
    }
    dst[((size_t)(b * HH + h) * T + i) * 64 + d] = val * scale;
}

__global__ __launch_bounds__(256) void gate_combine_kernel(
    const float* __restrict__ st, const float* __restrict__ f,
    const float* __restrict__ z, const float* __restrict__ ig,
    float* __restrict__ out, int n) {
    int i = blockIdx.x * 256 + threadIdx.x;
    if (i < n) out[i] = st[i] * f[i] + z[i] * ig[i];
}

// Pack f32 weight [K,N] row-major into WMMA-native bf16 slab [K/32][N][32].
__global__ __launch_bounds__(256) void pack_b_kernel(
    const float* __restrict__ W, bf16_t* __restrict__ out, int N, int total) {
    int idx = blockIdx.x * 256 + threadIdx.x;
    if (idx >= total) return;
    int kk   = idx & 31;
    int rest = idx >> 5;
    int n    = rest % N;
    int kt   = rest / N;
    out[idx] = f2b(W[(size_t)(kt * 32 + kk) * N + n]);
}

// ---------------------------------------------------------------------------
// Host-side orchestration
// ---------------------------------------------------------------------------
static inline void launch_gemm(hipStream_t s, int act, bool hb, bool ha,
                               const float* A, const bf16_t* B, float* C,
                               const float* bias, const float* add,
                               int M, int N, int K) {
    dim3 g(N / 128, M / 128), blk(256);
#define GEMM_CASE(a, hbv, hav) \
    gemm_bf16_kernel<a, hbv, hav><<<g, blk, 0, s>>>(A, B, C, bias, add, M, N, K)
    if (act == 0 && !hb && !ha)      GEMM_CASE(0, false, false);
    else if (act == 0 && hb && !ha)  GEMM_CASE(0, true, false);
    else if (act == 0 && !hb && ha)  GEMM_CASE(0, false, true);
    else if (act == 0 && hb && ha)   GEMM_CASE(0, true, true);
    else if (act == 1)               GEMM_CASE(1, true, false);
    else if (act == 2)               GEMM_CASE(2, true, false);
    else if (act == 3)               GEMM_CASE(3, true, false);
    else                             GEMM_CASE(4, true, false);
#undef GEMM_CASE
}

struct AttnW {
    const float* g; const bf16_t* wq; const bf16_t* wkv;
    const bf16_t* wo; const float* bo;
};

static void run_attention(hipStream_t s, const float* X, int Tq,
                          const float* ctx, int Tk, bool causal, bool rope,
                          const AttnW& w, float* out,
                          float* xn, float* qb, float* kvb,
                          float* qh, float* kh, float* vh, float* att) {
    const int BTq = BB * Tq, BTk = BB * Tk;
    rmsnorm_kernel<<<BTq, 256, 0, s>>>(X, w.g, xn);
    launch_gemm(s, 0, false, false, xn, w.wq, qb, nullptr, nullptr,
                BTq, INNER, DD);
    const float* kvin = ctx ? ctx : xn;
    launch_gemm(s, 0, false, false, kvin, w.wkv, kvb, nullptr, nullptr,
                BTk, 2 * INNER, DD);
    int tq = BB * HH * Tq * 64, tk = BB * HH * Tk * 64;
    prep_heads_kernel<<<tq / 256, 256, 0, s>>>(qb, INNER, qh, Tq,
                                               QK_SCALE, rope ? 1 : 0, tq);
    prep_heads_kernel<<<tk / 256, 256, 0, s>>>(kvb, 2 * INNER, kh, Tk,
                                               1.0f, rope ? 1 : 0, tk);
    prep_heads_kernel<<<tk / 256, 256, 0, s>>>(kvb + INNER, 2 * INNER, vh, Tk,
                                               1.0f, 0, tk);
    dim3 fg(Tq / 64, BB * HH);
    if (causal) flash_kernel<true><<<fg, 128, 0, s>>>(qh, kh, vh, att, Tq, Tk);
    else        flash_kernel<false><<<fg, 128, 0, s>>>(qh, kh, vh, att, Tq, Tk);
    launch_gemm(s, 0, true, false, att, w.wo, out, w.bo, nullptr,
                BTq, DD, INNER);
}

extern "C" void kernel_launch(void* const* d_in, const int* in_sizes, int n_in,
                              void* d_out, int out_size, void* d_ws,
                              size_t ws_size, hipStream_t stream) {
    (void)in_sizes; (void)n_in; (void)out_size; (void)ws_size;

    // ---- inputs (setup_inputs dict order, nested-dict insertion order) ----
    const float* x     = (const float*)d_in[0];   // [B,N,D]
    const float* state = (const float*)d_in[1];   // [B,S,D]
    const int attn_base[4] = {2, 7, 12, 17};      // input_self, state_self, in_cross, st_cross
    const float* pg_wm = (const float*)d_in[22];
    const float* pg_bm = (const float*)d_in[23];
    const float* pg_wi = (const float*)d_in[24];
    const float* pg_bi = (const float*)d_in[25];
    const float* pg_wf = (const float*)d_in[26];
    const float* pg_bf = (const float*)d_in[27];
    const float* fg_wm = (const float*)d_in[28];
    const float* fg_bm = (const float*)d_in[29];
    const float* fg_wi = (const float*)d_in[30];
    const float* fg_bi = (const float*)d_in[31];
    const float* fg_wf = (const float*)d_in[32];
    const float* fg_bf = (const float*)d_in[33];
    const float* in_proj_w = (const float*)d_in[34];  // [2D, D]
    const float* st_proj_w = (const float*)d_in[35];
    const float* iff_w1 = (const float*)d_in[36];
    const float* iff_b1 = (const float*)d_in[37];
    const float* iff_w2 = (const float*)d_in[38];
    const float* iff_b2 = (const float*)d_in[39];
    const float* sff_w1 = (const float*)d_in[40];
    const float* sff_b1 = (const float*)d_in[41];
    const float* sff_w2 = (const float*)d_in[42];
    const float* sff_b2 = (const float*)d_in[43];

    float* out_final = (float*)d_out;                        // [B,N,D]
    float* out_state = (float*)d_out + (size_t)BB * NN * DD; // [B,S,D]

    // ---- workspace carve ----
    char* wp = (char*)d_ws;
    auto alloc = [&](size_t bytes) -> void* {
        void* r = (void*)wp;
        wp += (bytes + 255) & ~(size_t)255;
        return r;
    };
    // Pack f32 [K,N] -> bf16 [K/32][N][32]
    auto pack = [&](const float* src, int K, int N) -> bf16_t* {
        size_t n = (size_t)K * N;
        bf16_t* d = (bf16_t*)alloc(n * sizeof(bf16_t));
        pack_b_kernel<<<(int)((n + 255) / 256), 256, 0, stream>>>(src, d, N, (int)n);
        return d;
    };

    // packed bf16 weights
    AttnW aw[4];
    for (int i = 0; i < 4; ++i) {
        int b0 = attn_base[i];
        aw[i].g   = (const float*)d_in[b0 + 0];
        aw[i].wq  = pack((const float*)d_in[b0 + 1], DD, INNER);
        aw[i].wkv = pack((const float*)d_in[b0 + 2], DD, 2 * INNER);
        aw[i].wo  = pack((const float*)d_in[b0 + 3], INNER, DD);
        aw[i].bo  = (const float*)d_in[b0 + 4];
    }
    bf16_t* pg_wm_b = pack(pg_wm, DD, DD);
    bf16_t* pg_wi_b = pack(pg_wi, DD, DD);
    bf16_t* pg_wf_b = pack(pg_wf, DD, DD);
    bf16_t* fg_wm_b = pack(fg_wm, DD, DD);
    bf16_t* fg_wi_b = pack(fg_wi, DD, DD);
    bf16_t* fg_wf_b = pack(fg_wf, DD, DD);
    bf16_t* ipw_top = pack(in_proj_w, DD, DD);                       // rows 0..D-1
    bf16_t* ipw_bot = pack(in_proj_w + (size_t)DD * DD, DD, DD);     // rows D..2D-1
    bf16_t* spw_top = pack(st_proj_w, DD, DD);
    bf16_t* spw_bot = pack(st_proj_w + (size_t)DD * DD, DD, DD);
    bf16_t* iw1_b   = pack(iff_w1, DD, 4 * DD);
    bf16_t* iw2_b   = pack(iff_w2, 4 * DD, DD);
    bf16_t* sw1_b   = pack(sff_w1, DD, 4 * DD);
    bf16_t* sw2_b   = pack(sff_w2, 4 * DD, DD);

    // f32 scratch
    const size_t TI = (size_t)BB * NN;       // 4096 input tokens
    const size_t TS = (size_t)BB * SS;       // 2048 state tokens
    float* xn   = (float*)alloc(TI * DD * 4);
    float* qb   = (float*)alloc(TI * INNER * 4);
    float* kvb  = (float*)alloc(TI * 2 * INNER * 4);
    float* qh   = (float*)alloc(TI * INNER * 4);
    float* kh   = (float*)alloc(TI * INNER * 4);
    float* vh   = (float*)alloc(TI * INNER * 4);
    float* att  = (float*)alloc(TI * INNER * 4);
    float* input_attn = (float*)alloc(TI * DD * 4);
    float* state_attn = (float*)alloc(TS * DD * 4);
    float* in_cross_o = (float*)alloc(TI * DD * 4);
    float* st_cross_o = (float*)alloc(TS * DD * 4);
    float* input_res  = (float*)alloc(TI * DD * 4);
    float* proj_st    = (float*)alloc(TS * DD * 4);
    float* state_res  = (float*)alloc(TS * DD * 4);
    float* zb   = (float*)alloc(TS * DD * 4);
    float* ib   = (float*)alloc(TS * DD * 4);
    float* fb   = (float*)alloc(TS * DD * 4);
    float* hbig = (float*)alloc(TI * 4 * DD * 4);
    float* ffs  = (float*)alloc(TS * DD * 4);

    // ---- 4 attention blocks ----
    run_attention(stream, x, NN, nullptr, NN, true,  true,  aw[0], input_attn,
                  xn, qb, kvb, qh, kh, vh, att);
    run_attention(stream, state, SS, nullptr, SS, false, true,  aw[1], state_attn,
                  xn, qb, kvb, qh, kh, vh, att);
    run_attention(stream, x, NN, state, SS, false, false, aw[2], in_cross_o,
                  xn, qb, kvb, qh, kh, vh, att);
    run_attention(stream, state, SS, x, NN, false, false, aw[3], st_cross_o,
                  xn, qb, kvb, qh, kh, vh, att);

    // ---- projections (concat expressed as two accumulating GEMMs) ----
    // input_residual = in_cross@Wtop + input_attn@Wbot + x
    launch_gemm(stream, 0, false, true, in_cross_o, ipw_top, input_res,
                nullptr, x, (int)TI, DD, DD);
    launch_gemm(stream, 0, false, true, input_attn, ipw_bot,
                input_res, nullptr, input_res, (int)TI, DD, DD);
    // proj_st = st_cross@Wtop + state_attn@Wbot
    launch_gemm(stream, 0, false, false, st_cross_o, spw_top, proj_st,
                nullptr, nullptr, (int)TS, DD, DD);
    launch_gemm(stream, 0, false, true, state_attn, spw_bot,
                proj_st, nullptr, proj_st, (int)TS, DD, DD);

    // ---- state_residual = gate(proj_st, state, proj_gate) ----
    launch_gemm(stream, 2, true, false, proj_st, pg_wm_b, zb, pg_bm, nullptr,
                (int)TS, DD, DD);
    launch_gemm(stream, 3, true, false, proj_st, pg_wi_b, ib, pg_bi, nullptr,
                (int)TS, DD, DD);
    launch_gemm(stream, 4, true, false, proj_st, pg_wf_b, fb, pg_bf, nullptr,
                (int)TS, DD, DD);
    {
        int n = (int)(TS * DD);
        gate_combine_kernel<<<(n + 255) / 256, 256, 0, stream>>>(
            state, fb, zb, ib, state_res, n);
    }

    // ---- output = ff(input_residual) + input_residual ----
    launch_gemm(stream, 1, true, false, input_res, iw1_b, hbig, iff_b1, nullptr,
                (int)TI, 4 * DD, DD);
    launch_gemm(stream, 0, true, true, hbig, iw2_b, out_final, iff_b2,
                input_res, (int)TI, DD, 4 * DD);

    // ---- next_state = gate(ff(state_residual), state_residual, ff_gate) ----
    launch_gemm(stream, 1, true, false, state_res, sw1_b, hbig, sff_b1, nullptr,
                (int)TS, 4 * DD, DD);
    launch_gemm(stream, 0, true, false, hbig, sw2_b, ffs, sff_b2, nullptr,
                (int)TS, DD, 4 * DD);
    launch_gemm(stream, 2, true, false, ffs, fg_wm_b, zb, fg_bm, nullptr,
                (int)TS, DD, DD);
    launch_gemm(stream, 3, true, false, ffs, fg_wi_b, ib, fg_bi, nullptr,
                (int)TS, DD, DD);
    launch_gemm(stream, 4, true, false, ffs, fg_wf_b, fb, fg_bf, nullptr,
                (int)TS, DD, DD);
    {
        int n = (int)(TS * DD);
        gate_combine_kernel<<<(n + 255) / 256, 256, 0, stream>>>(
            state_res, fb, zb, ib, out_state, n);
    }
}
